// AttentionWPooling_54125177864330
// MI455X (gfx1250) — matrix operand order repaired and checked by reading, not compiled, not run on hardware
//
#include <hip/hip_runtime.h>
#include <hip/hip_bf16.h>

// ---------------------------------------------------------------------------
// AttentionWPooling on gfx1250 (MI455X), wave32 + V_WMMA_F32_16X16X32_BF16.
//
// cross = a.b^T done with bf16 hi/lo split (3 WMMAs) -> ~fp32 accuracy on the
// bf16 tensor pipe. A = 1/(1+sqrt(max(sq_i+sq_j-2*cross,0))) is reduced to
// row/col sums on the fly (the 2050x2050x32 matrix is never materialized).
// ---------------------------------------------------------------------------

typedef __attribute__((ext_vector_type(16))) __bf16 v16bf;
typedef __attribute__((ext_vector_type(8)))  float  v8f;
typedef __attribute__((ext_vector_type(4)))  float  vf4;

#define S_TOT 2050            // 2048 + 3 - 1
#define L_OUT 2048
#define H_DIM 128
#define N_B   32
#define JT    129             // ceil(2050/16) j-tiles
#define SP    (JT * 16)       // 2064 padded sequence length in workspace
#define IBLK  64              // i-rows per block (4 waves x 16)
#define NIB   33              // ceil(2050/64) i-blocks
#define LDSW  136             // padded LDS row stride (bf16 elems), 16B-aligned

// ---------------------------------------------------------------------------
// Kernel 1: squared norms, padded region written as 0.
// ---------------------------------------------------------------------------
__global__ __launch_bounds__(128)
void sq_norms(const float* __restrict__ x0, const float* __restrict__ x1,
              float* __restrict__ sq0, float* __restrict__ sq1)
{
    const int s = blockIdx.x;          // 0..SP-1
    const int b = blockIdx.y;
    const int h = threadIdx.x;         // 0..127
    float v0 = 0.0f, v1 = 0.0f;
    if (s < S_TOT) {
        float a = x0[((size_t)b * S_TOT + s) * H_DIM + h];
        float c = x1[((size_t)b * S_TOT + s) * H_DIM + h];
        v0 = a * a;
        v1 = c * c;
    }
#pragma unroll
    for (int m = 1; m < 32; m <<= 1) {
        v0 += __shfl_xor(v0, m, 32);
        v1 += __shfl_xor(v1, m, 32);
    }
    __shared__ float sm[8];
    const int wave = h >> 5, lane = h & 31;
    if (lane == 0) { sm[wave] = v0; sm[4 + wave] = v1; }
    __syncthreads();
    if (h == 0) {
        sq0[b * SP + s] = sm[0] + sm[1] + sm[2] + sm[3];
        sq1[b * SP + s] = sm[4] + sm[5] + sm[6] + sm[7];
    }
}

// ---------------------------------------------------------------------------
// Kernel 2: WMMA core. Block = 128 threads (4 waves), owns 64 i-rows; each
// wave owns a 16-row tile and sweeps all 129 j-tiles. B-tile staged in LDS
// as bf16 hi/lo once per block, shared by all 4 waves.
// ---------------------------------------------------------------------------
__global__ __launch_bounds__(128)
void pair_sums(const float* __restrict__ x0, const float* __restrict__ x1,
               const float* __restrict__ sq0, const float* __restrict__ sq1,
               float* __restrict__ colsum, float* __restrict__ rowsum)
{
    __shared__ __bf16 lds_bh[16 * LDSW];
    __shared__ __bf16 lds_bl[16 * LDSW];

    const int iblk = blockIdx.x;
    const int b    = blockIdx.y;
    const int tid  = threadIdx.x;
    const int wave = tid >> 5;
    const int lane = tid & 31;
    const int g    = lane >> 4;        // half-wave group
    const int l16  = lane & 15;

    const float* Am   = x0 + (size_t)b * S_TOT * H_DIM;
    const float* Bm   = x1 + (size_t)b * S_TOT * H_DIM;
    const float* sq0b = sq0 + b * SP;
    const float* sq1b = sq1 + b * SP;

    const int i0 = iblk * IBLK + wave * 16;      // wave's 16-row i-tile base

    // ---- preload A fragments, fp32 -> bf16 hi + residual lo ----
    // A fragment k order used here (16x32, MxK): lane = M (mod 16); half-wave g=0
    // holds K {0..15}, g=1 holds K {16..31} per 32-K block.
    const int arow = i0 + l16;
    const float* arowp = Am + (size_t)(arow < S_TOT ? arow : 0) * H_DIM;
    v16bf ah[4], al[4];
#pragma unroll
    for (int kb = 0; kb < 4; ++kb) {
        const float* p = arowp + kb * 32 + 16 * g;
        vf4 q0 = ((const vf4*)p)[0];
        vf4 q1 = ((const vf4*)p)[1];
        vf4 q2 = ((const vf4*)p)[2];
        vf4 q3 = ((const vf4*)p)[3];
        float t[16];
        *(vf4*)&t[0] = q0; *(vf4*)&t[4] = q1; *(vf4*)&t[8] = q2; *(vf4*)&t[12] = q3;
#pragma unroll
        for (int e = 0; e < 16; ++e) {
            __bf16 hi = (__bf16)t[e];
            ah[kb][e] = hi;
            al[kb][e] = (__bf16)(t[e] - (float)hi);
        }
    }

    float rs[8];
#pragma unroll
    for (int v = 0; v < 8; ++v) rs[v] = 0.0f;

    // cooperative staging: 128 threads x 16 elems = 16 rows x 128 cols
    const int srow = tid >> 3;
    const int scol = (tid & 7) * 16;

    for (int jt = 0; jt < JT; ++jt) {
        const int j0 = jt * 16;

        __syncthreads();   // previous tile's ds reads complete
        {
            const int brow = j0 + srow;
            const float* p = Bm + (size_t)(brow < S_TOT ? brow : 0) * H_DIM + scol;
            vf4 q0 = ((const vf4*)p)[0];
            vf4 q1 = ((const vf4*)p)[1];
            vf4 q2 = ((const vf4*)p)[2];
            vf4 q3 = ((const vf4*)p)[3];
            float t[16];
            *(vf4*)&t[0] = q0; *(vf4*)&t[4] = q1; *(vf4*)&t[8] = q2; *(vf4*)&t[12] = q3;
            __bf16* dh = &lds_bh[srow * LDSW + scol];
            __bf16* dl = &lds_bl[srow * LDSW + scol];
#pragma unroll
            for (int e = 0; e < 16; ++e) {
                __bf16 hi = (__bf16)t[e];
                dh[e] = hi;
                dl[e] = (__bf16)(t[e] - (float)hi);
            }
        }
        __syncthreads();

        // B fragment k order used here, same as A (32x16, KxN): lane = N (mod 16); half-wave
        // g=0 holds K {0..15}, g=1 holds K {16..31} per 32-K block -> one
        // contiguous 32-byte run per fragment.
        v8f acc = {};
#pragma unroll
        for (int kb = 0; kb < 4; ++kb) {
            const int koff = kb * 32 + 16 * g;
            v16bf bh = *(const v16bf*)&lds_bh[l16 * LDSW + koff];
            v16bf bl = *(const v16bf*)&lds_bl[l16 * LDSW + koff];
            acc = __builtin_amdgcn_wmma_f32_16x16x32_bf16(false, ah[kb], false, bh,
                                                          (short)0, acc, false, false);
            acc = __builtin_amdgcn_wmma_f32_16x16x32_bf16(false, ah[kb], false, bl,
                                                          (short)0, acc, false, false);
            acc = __builtin_amdgcn_wmma_f32_16x16x32_bf16(false, al[kb], false, bh,
                                                          (short)0, acc, false, false);
        }

        // transform + on-the-fly reductions. C layout: VGPR v, lane ->
        // M = v + 8*g, N = l16.
        const int   jv = j0 + l16;
        const float s1 = sq1b[jv];
        float cs = 0.0f;
#pragma unroll
        for (int v = 0; v < 8; ++v) {
            const int iv = i0 + v + 8 * g;
            const int ic = iv < SP ? iv : (SP - 1);
            float d2 = sq0b[ic] + s1 - 2.0f * acc[v];
            d2 = fmaxf(d2, 0.0f);
            float Aij = __builtin_amdgcn_rcpf(1.0f + __builtin_amdgcn_sqrtf(d2));
            Aij = (iv < S_TOT && jv < S_TOT) ? Aij : 0.0f;
            rs[v] += Aij;
            cs += Aij;
        }
        cs += __shfl_xor(cs, 16, 32);            // combine M halves (same N)
        if (lane < 16 && jv < S_TOT)
            atomicAdd(&colsum[b * SP + jv], cs); // r[j] = sum_i A[i,j]
    }

    // row sums c[i] = sum_j A[i,j]: reduce across the 16 N-lanes, plain store
    // (i-rows are partitioned across blocks/waves).
#pragma unroll
    for (int v = 0; v < 8; ++v) {
        float r = rs[v];
        r += __shfl_xor(r, 1, 32);
        r += __shfl_xor(r, 2, 32);
        r += __shfl_xor(r, 4, 32);
        r += __shfl_xor(r, 8, 32);
        const int iv = i0 + v + 8 * g;
        if (l16 == 0 && iv < S_TOT)
            rowsum[b * SP + iv] = r;
    }
}

// ---------------------------------------------------------------------------
// Kernel 3: wp0[t] = sum_{d=0..2} r[t+d]*a[t+d], wp1 likewise with c and x1.
// ---------------------------------------------------------------------------
__global__ __launch_bounds__(128)
void window_pool(const float* __restrict__ x0, const float* __restrict__ x1,
                 const float* __restrict__ colsum, const float* __restrict__ rowsum,
                 float* __restrict__ out0, float* __restrict__ out1)
{
    const int t = blockIdx.x;          // 0..2047
    const int b = blockIdx.y;
    const int h = threadIdx.x;         // 0..127
    const float* a  = x0 + (size_t)b * S_TOT * H_DIM;
    const float* bb = x1 + (size_t)b * S_TOT * H_DIM;
    const float* r  = colsum + b * SP;
    const float* c  = rowsum + b * SP;
    float o0 = r[t]     * a[(size_t)t       * H_DIM + h]
             + r[t + 1] * a[(size_t)(t + 1) * H_DIM + h]
             + r[t + 2] * a[(size_t)(t + 2) * H_DIM + h];
    float o1 = c[t]     * bb[(size_t)t       * H_DIM + h]
             + c[t + 1] * bb[(size_t)(t + 1) * H_DIM + h]
             + c[t + 2] * bb[(size_t)(t + 2) * H_DIM + h];
    out0[((size_t)b * L_OUT + t) * H_DIM + h] = o0;
    out1[((size_t)b * L_OUT + t) * H_DIM + h] = o1;
}

// ---------------------------------------------------------------------------
extern "C" void kernel_launch(void* const* d_in, const int* in_sizes, int n_in,
                              void* d_out, int out_size, void* d_ws, size_t ws_size,
                              hipStream_t stream)
{
    (void)in_sizes; (void)n_in; (void)out_size; (void)ws_size;

    const float* x0 = (const float*)d_in[0];
    const float* x1 = (const float*)d_in[1];
    float* out0 = (float*)d_out;
    float* out1 = out0 + (size_t)N_B * L_OUT * H_DIM;

    float* sq0    = (float*)d_ws;              // 32*2064 f32 each
    float* sq1    = sq0 + (size_t)N_B * SP;
    float* colsum = sq1 + (size_t)N_B * SP;
    float* rowsum = colsum + (size_t)N_B * SP;

    hipMemsetAsync(colsum, 0, (size_t)N_B * SP * sizeof(float), stream);
    sq_norms  <<<dim3(SP,    N_B), 128, 0, stream>>>(x0, x1, sq0, sq1);
    pair_sums <<<dim3(NIB,   N_B), 128, 0, stream>>>(x0, x1, sq0, sq1, colsum, rowsum);
    window_pool<<<dim3(L_OUT, N_B), 128, 0, stream>>>(x0, x1, colsum, rowsum, out0, out1);
}
